// GVPGraphormerAttention_38689065402611
// MI455X (gfx1250) — compile-verified, matrix operand order checked by reading, not run on hardware
//
#include <hip/hip_runtime.h>
#include <hip/hip_bf16.h>
#include <math.h>

// ---------------------------------------------------------------------------
// GVP-Graphormer attention for MI455X (gfx1250, wave32, WMMA).
// Flash-style two-pass softmax; QK^T and alpha@V run on v_wmma_f32_16x16x32_f16.
// ---------------------------------------------------------------------------

typedef __attribute__((ext_vector_type(16))) _Float16 v16h;
typedef __attribute__((ext_vector_type(8)))  _Float16 v8h;
typedef __attribute__((ext_vector_type(8)))  float    v8f;

#define N_NODES 1024
#define C_CONF  8
#define SI      128
#define VI      16
#define NH      4          // heads
#define HS      32         // scalar feats per head (128/4)
#define HV      4          // vector feats per head (16/4)
#define KF      44         // 32 + 3*4 packed features per head
#define KP      64         // padded K for two 16x16x32 WMMA k-blocks
#define NPLANE  (C_CONF*NH)
#define E_EDGES 16384
#define MAXD    5

__device__ __forceinline__ float sigf(float x) { return 1.0f / (1.0f + __expf(-x)); }

// Load one lane's A/B fragment half (16 f16) for v_wmma_f32_16x16x32_f16.
// Per the CDNA5 16-bit A/B layout: elements 0..7 = K[base .. base+7],
// elements 8..15 = K[base+16 .. base+23]; caller pre-offsets base by hi*8.
__device__ __forceinline__ v16h frag_ld(const _Float16* p) {
  v8h lo = *(const v8h*)(p);
  v8h hi = *(const v8h*)(p + 16);
  v16h r;
#pragma unroll
  for (int i = 0; i < 8; ++i) { r[i] = lo[i]; r[i + 8] = hi[i]; }
  return r;
}

// ---------------------------------------------------------------------------
// Stage 0: bias[h][i][j] = spatial_emb[min(spatial[i][j],5)][h]
// ---------------------------------------------------------------------------
__global__ void bias_build_kernel(const int* __restrict__ spatial,
                                  const float* __restrict__ emb,
                                  float* __restrict__ bias) {
  int idx = blockIdx.x * blockDim.x + threadIdx.x;   // over N*N
  int sp = spatial[idx]; sp = sp > MAXD ? MAXD : sp;
  size_t NN = (size_t)N_NODES * N_NODES;
#pragma unroll
  for (int h = 0; h < NH; ++h)
    bias[(size_t)h * NN + idx] = emb[sp * NH + h];
}

// ---------------------------------------------------------------------------
// Stage 0b: edge encoder (gvp_scalar_out) + symmetric scatter into bias.
// bias[h][a][b] = spatial_term(a,b) + eb[h]  (overwrite, matching .set)
// ---------------------------------------------------------------------------
__global__ void edge_bias_kernel(const int* __restrict__ eidx,
                                 const float* __restrict__ eas,
                                 const float* __restrict__ eav,
                                 const float* __restrict__ ewh,
                                 const float* __restrict__ esw,
                                 const float* __restrict__ esb,
                                 const int* __restrict__ spatial,
                                 const float* __restrict__ emb,
                                 float* __restrict__ bias) {
  int e = blockIdx.x * blockDim.x + threadIdx.x;
  if (e >= E_EDGES) return;
  // vh[d][hh] = sum_k eav[e][k][d] * ewh[k][hh]
  float vh[3][4];
#pragma unroll
  for (int d = 0; d < 3; ++d)
#pragma unroll
    for (int hh = 0; hh < 4; ++hh) {
      float a = 0.f;
#pragma unroll
      for (int k = 0; k < 4; ++k) a += eav[(e * 4 + k) * 3 + d] * ewh[k * 4 + hh];
      vh[d][hh] = a;
    }
  float vn[4];
#pragma unroll
  for (int k = 0; k < 4; ++k)
    vn[k] = sqrtf(fmaxf(vh[0][k]*vh[0][k] + vh[1][k]*vh[1][k] + vh[2][k]*vh[2][k], 1e-8f));
  float eb[NH];
#pragma unroll
  for (int h = 0; h < NH; ++h) {
    float a = esb[h];
    for (int k = 0; k < 32; ++k) a += eas[e * 32 + k] * esw[k * NH + h];
#pragma unroll
    for (int k = 0; k < 4; ++k)  a += vn[k] * esw[(32 + k) * NH + h];
    eb[h] = a;
  }
  int s = eidx[e], d = eidx[E_EDGES + e];
  size_t NN = (size_t)N_NODES * N_NODES;
  int sp1 = spatial[s * N_NODES + d]; sp1 = sp1 > MAXD ? MAXD : sp1;
  int sp2 = spatial[d * N_NODES + s]; sp2 = sp2 > MAXD ? MAXD : sp2;
#pragma unroll
  for (int h = 0; h < NH; ++h) {
    bias[(size_t)h * NN + (size_t)s * N_NODES + d] = emb[sp1 * NH + h] + eb[h];
    bias[(size_t)h * NN + (size_t)d * N_NODES + s] = emb[sp2 * NH + h] + eb[h];
  }
}

// ---------------------------------------------------------------------------
// Shared GVP core (one (node,conf) row per 128-thread block).
// Produces vh[3*16], vn[16], s_out[128], sg=sigmoid(s_out), gate[16] in LDS.
// ---------------------------------------------------------------------------
__device__ __forceinline__ void gvp_core(int tid,
    const float* __restrict__ wh, const float* __restrict__ sw,
    const float* __restrict__ sb, const float* __restrict__ gw,
    const float* __restrict__ gb,
    const float* s_in, const float* v_in,
    float* vh, float* vn, float* s_out, float* sg, float* gate) {
  if (tid < 48) {                       // vh[d][hh] = sum_k v[k][d]*wh[k][hh]
    int d = tid >> 4, hh = tid & 15;
    float a = 0.f;
#pragma unroll
    for (int k = 0; k < VI; ++k) a += v_in[k * 3 + d] * wh[k * VI + hh];
    vh[d * VI + hh] = a;
  }
  __syncthreads();
  if (tid < VI) {                       // vn = sqrt(clip(sum_d vh^2, eps))
    float a = vh[tid]*vh[tid] + vh[VI+tid]*vh[VI+tid] + vh[2*VI+tid]*vh[2*VI+tid];
    vn[tid] = sqrtf(fmaxf(a, 1e-8f));
  }
  __syncthreads();
  {                                     // s_out[j] = [s|vn] @ ws_w + b
    float a = sb[tid];
    for (int k = 0; k < SI; ++k) a += s_in[k] * sw[k * SI + tid];
#pragma unroll
    for (int k = 0; k < VI; ++k) a += vn[k] * sw[(SI + k) * SI + tid];
    s_out[tid] = a;
    sg[tid] = sigf(a);
  }
  __syncthreads();
  if (tid < VI) {                       // gate = sigmoid(sigmoid(s_out)@wsv + b)
    float a = gb[tid];
    for (int j = 0; j < SI; ++j) a += sg[j] * gw[j * VI + tid];
    gate[tid] = sigf(a);
  }
  __syncthreads();
}

// ---------------------------------------------------------------------------
// Stage 1: q/k/v GVPs + pack features to f16.
// mode 0: Qp [plane][n][64], scaled by 1/6 (folds 1/sqrt(hs+hv))
// mode 1: Kp [plane][n][64]
// mode 2: Vt [plane][64][n]   (transposed for B-fragment loads of alpha@V)
// ---------------------------------------------------------------------------
__global__ void gvp_pack_kernel(const float* __restrict__ xs,
                                const float* __restrict__ xv,
                                const float* __restrict__ wh,
                                const float* __restrict__ sw,
                                const float* __restrict__ sb,
                                const float* __restrict__ wv,
                                const float* __restrict__ gw,
                                const float* __restrict__ gb,
                                _Float16* __restrict__ outp, int mode) {
  __shared__ float s_in[SI], v_in[VI * 3], vh[3 * VI], vn[VI];
  __shared__ float s_out[SI], sg[SI], gate[VI], vout[VI * 3];
  int tid = threadIdx.x;
  int n = blockIdx.x / C_CONF, c = blockIdx.x % C_CONF;
  const float* xsrow = xs + ((size_t)n * C_CONF + c) * SI;
  const float* xvrow = xv + ((size_t)n * C_CONF + c) * VI * 3;
  if (tid < SI)     s_in[tid] = xsrow[tid];
  if (tid < VI * 3) v_in[tid] = xvrow[tid];
  __syncthreads();
  gvp_core(tid, wh, sw, sb, gw, gb, s_in, v_in, vh, vn, s_out, sg, gate);
  if (tid < 48) {                       // v_out[u][d] = (sum_hh vh[d][hh]*wv[hh][u]) * gate[u]
    int u = tid / 3, d = tid % 3;
    float a = 0.f;
#pragma unroll
    for (int hh = 0; hh < VI; ++hh) a += vh[d * VI + hh] * wv[hh * VI + u];
    vout[u * 3 + d] = a * gate[u];
  }
  __syncthreads();
  float scale = (mode == 0) ? (1.0f / 6.0f) : 1.0f;
  for (int slot = tid; slot < NH * KP; slot += blockDim.x) {
    int h = slot / KP, kk = slot % KP;
    float val;
    if (kk < HS)      val = s_out[h * HS + kk] * sg[h * HS + kk];   // silu
    else if (kk < KF) { int r = kk - HS; val = vout[(h * HV + r / 3) * 3 + r % 3]; }
    else              val = 0.f;                                    // pad
    val *= scale;
    int plane = c * NH + h;
    if (mode == 2) outp[((size_t)plane * KP + kk) * N_NODES + n] = (_Float16)val;
    else           outp[((size_t)plane * N_NODES + n) * KP + kk] = (_Float16)val;
  }
}

// ---------------------------------------------------------------------------
// Stage 2 (pass A): per (plane, 16-row i-tile) online softmax stats over j.
// One wave per block; QK^T via two chained v_wmma_f32_16x16x32_f16.
// ---------------------------------------------------------------------------
__global__ void attn_stats_kernel(const _Float16* __restrict__ Qp,
                                  const _Float16* __restrict__ Kp,
                                  const float* __restrict__ bias,
                                  float* __restrict__ partM,
                                  float* __restrict__ partS) {
  int lane = threadIdx.x;
  int r = lane & 15, hi = lane >> 4;
  int i0 = blockIdx.x * 16;
  int plane = blockIdx.y;
  int h = plane & (NH - 1);
  const _Float16* Qpl = Qp + (size_t)plane * N_NODES * KP;
  const _Float16* Kpl = Kp + (size_t)plane * N_NODES * KP;
  const float* bh = bias + (size_t)h * N_NODES * N_NODES;

  v16h a0 = frag_ld(Qpl + (size_t)(i0 + r) * KP + 0  + hi * 8);
  v16h a1 = frag_ld(Qpl + (size_t)(i0 + r) * KP + 32 + hi * 8);

  float mrun[8], srun[8];
#pragma unroll
  for (int v = 0; v < 8; ++v) { mrun[v] = -1e30f; srun[v] = 0.f; }

  for (int j0 = 0; j0 < N_NODES; j0 += 16) {
    v16h b0 = frag_ld(Kpl + (size_t)(j0 + r) * KP + 0  + hi * 8);
    v16h b1 = frag_ld(Kpl + (size_t)(j0 + r) * KP + 32 + hi * 8);
    v8f acc = {};
    acc = __builtin_amdgcn_wmma_f32_16x16x32_f16(false, a0, false, b0, (short)0, acc, false, false);
    acc = __builtin_amdgcn_wmma_f32_16x16x32_f16(false, a1, false, b1, (short)0, acc, false, false);
    if (j0 + 16 < N_NODES)
      __builtin_prefetch(Kpl + (size_t)(j0 + 16 + r) * KP, 0, 1);
#pragma unroll
    for (int v = 0; v < 8; ++v) {       // D layout: M = v + 8*hi, N = lane&15
      int m = v + 8 * hi;
      float e = acc[v] + bh[(size_t)(i0 + m) * N_NODES + j0 + r];
      float mn = fmaxf(mrun[v], e);
      srun[v] = srun[v] * __expf(mrun[v] - mn) + __expf(e - mn);
      mrun[v] = mn;
    }
  }
  // combine per-row across the 16 lanes that share each row
  __shared__ float redM[16][17], redS[16][17];
#pragma unroll
  for (int v = 0; v < 8; ++v) { redM[v + 8 * hi][r] = mrun[v]; redS[v + 8 * hi][r] = srun[v]; }
  __syncthreads();
  if (lane < 16) {
    float m = -1e30f;
#pragma unroll
    for (int k = 0; k < 16; ++k) m = fmaxf(m, redM[lane][k]);
    float s = 0.f;
#pragma unroll
    for (int k = 0; k < 16; ++k) s += redS[lane][k] * __expf(redM[lane][k] - m);
    partM[(size_t)plane * N_NODES + i0 + lane] = m;
    partS[(size_t)plane * N_NODES + i0 + lane] = s;
  }
}

// ---------------------------------------------------------------------------
// Stage 3: reduce the 32 per-plane partial stats into global rowM / 1/rowS.
// ---------------------------------------------------------------------------
__global__ void softmax_combine_kernel(const float* __restrict__ partM,
                                       const float* __restrict__ partS,
                                       float* __restrict__ rowM,
                                       float* __restrict__ rowInvS) {
  int i = blockIdx.x * blockDim.x + threadIdx.x;
  if (i >= N_NODES) return;
  float m = -1e30f;
  for (int p = 0; p < NPLANE; ++p) m = fmaxf(m, partM[(size_t)p * N_NODES + i]);
  float s = 0.f;
  for (int p = 0; p < NPLANE; ++p)
    s += partS[(size_t)p * N_NODES + i] * __expf(partM[(size_t)p * N_NODES + i] - m);
  rowM[i] = m;
  rowInvS[i] = 1.0f / s;
}

// ---------------------------------------------------------------------------
// Stage 4 (pass B): recompute QK tiles, normalize (alpha), and accumulate
// alpha @ V with WMMA. Alpha D-fragments are restaged through a 1KB LDS tile
// to re-load them in A-fragment layout.
// ---------------------------------------------------------------------------
__global__ void attn_av_kernel(const _Float16* __restrict__ Qp,
                               const _Float16* __restrict__ Kp,
                               const _Float16* __restrict__ Vt,
                               const float* __restrict__ bias,
                               const float* __restrict__ rowM,
                               const float* __restrict__ rowInvS,
                               float* __restrict__ Of) {
  __shared__ __attribute__((aligned(16))) _Float16 alphaT[16 * 32];
  int lane = threadIdx.x;
  int r = lane & 15, hi = lane >> 4;
  int i0 = blockIdx.x * 16;
  int plane = blockIdx.y;
  int h = plane & (NH - 1);
  const _Float16* Qpl = Qp + (size_t)plane * N_NODES * KP;
  const _Float16* Kpl = Kp + (size_t)plane * N_NODES * KP;
  const _Float16* Vpl = Vt + (size_t)plane * KP * N_NODES;
  const float* bh = bias + (size_t)h * N_NODES * N_NODES;

  v16h a0 = frag_ld(Qpl + (size_t)(i0 + r) * KP + 0  + hi * 8);
  v16h a1 = frag_ld(Qpl + (size_t)(i0 + r) * KP + 32 + hi * 8);

  float rm[8], ri[8];
#pragma unroll
  for (int v = 0; v < 8; ++v) {
    int m = v + 8 * hi;
    rm[v] = rowM[i0 + m];
    ri[v] = rowInvS[i0 + m];
  }
  v8f zero = {};
  v8f acc4[4] = {zero, zero, zero, zero};

  for (int j0 = 0; j0 < N_NODES; j0 += 32) {
#pragma unroll
    for (int t = 0; t < 2; ++t) {
      int jt = j0 + t * 16;
      v16h b0 = frag_ld(Kpl + (size_t)(jt + r) * KP + 0  + hi * 8);
      v16h b1 = frag_ld(Kpl + (size_t)(jt + r) * KP + 32 + hi * 8);
      v8f acc = {};
      acc = __builtin_amdgcn_wmma_f32_16x16x32_f16(false, a0, false, b0, (short)0, acc, false, false);
      acc = __builtin_amdgcn_wmma_f32_16x16x32_f16(false, a1, false, b1, (short)0, acc, false, false);
#pragma unroll
      for (int v = 0; v < 8; ++v) {
        int m = v + 8 * hi;
        float e = acc[v] + bh[(size_t)(i0 + m) * N_NODES + jt + r];
        float p = __expf(e - rm[v]) * ri[v];
        alphaT[m * 32 + t * 16 + r] = (_Float16)p;
      }
    }
    __syncthreads();                    // LDS write -> cross-lane read
    v16h aa = frag_ld(&alphaT[r * 32 + hi * 8]);
    if (j0 + 32 < N_NODES)
      __builtin_prefetch(Kpl + (size_t)(j0 + 32 + r) * KP, 0, 1);
#pragma unroll
    for (int t2 = 0; t2 < 4; ++t2) {
      int f = t2 * 16 + r;              // B lane = output feature column
      v16h bv = frag_ld(Vpl + (size_t)f * N_NODES + j0 + hi * 8);
      acc4[t2] = __builtin_amdgcn_wmma_f32_16x16x32_f16(false, aa, false, bv,
                                                        (short)0, acc4[t2], false, false);
    }
    __syncthreads();                    // before alphaT is overwritten
  }
  float* Ofpl = Of + (size_t)plane * N_NODES * KP;
#pragma unroll
  for (int t2 = 0; t2 < 4; ++t2)
#pragma unroll
    for (int v = 0; v < 8; ++v)
      Ofpl[(size_t)(i0 + v + 8 * hi) * KP + t2 * 16 + r] = acc4[t2][v];
}

// ---------------------------------------------------------------------------
// Stage 5: gather attention output, run the output GVP, write final results.
// d_out = [ silu(s_out) : N*C*128 | v_out : N*C*16*3 ] (f32, flat)
// ---------------------------------------------------------------------------
__global__ void gvp_out_kernel(const float* __restrict__ Of,
                               const float* __restrict__ wh,
                               const float* __restrict__ sw,
                               const float* __restrict__ sb,
                               const float* __restrict__ wv,
                               const float* __restrict__ gw,
                               const float* __restrict__ gb,
                               float* __restrict__ out) {
  __shared__ float s_in[SI], v_in[VI * 3], vh[3 * VI], vn[VI];
  __shared__ float s_out[SI], sg[SI], gate[VI];
  int tid = threadIdx.x;
  int n = blockIdx.x / C_CONF, c = blockIdx.x % C_CONF;
  if (tid < SI) {
    int h = tid / HS, kk = tid % HS;
    s_in[tid] = Of[((size_t)(c * NH + h) * N_NODES + n) * KP + kk];
  }
  if (tid < VI * 3) {
    int u = tid / 3, d = tid % 3;
    int h = u / HV, uu = u % HV;
    v_in[tid] = Of[((size_t)(c * NH + h) * N_NODES + n) * KP + HS + uu * 3 + d];
  }
  __syncthreads();
  gvp_core(tid, wh, sw, sb, gw, gb, s_in, v_in, vh, vn, s_out, sg, gate);
  float* outs = out;
  float* outv = out + (size_t)N_NODES * C_CONF * SI;
  if (tid < SI)
    outs[((size_t)n * C_CONF + c) * SI + tid] = s_out[tid] * sg[tid];
  if (tid < 48) {
    int u = tid / 3, d = tid % 3;
    float a = 0.f;
#pragma unroll
    for (int hh = 0; hh < VI; ++hh) a += vh[d * VI + hh] * wv[hh * VI + u];
    outv[(((size_t)n * C_CONF + c) * VI + u) * 3 + d] = a * gate[u];
  }
}

// ---------------------------------------------------------------------------
extern "C" void kernel_launch(void* const* d_in, const int* in_sizes, int n_in,
                              void* d_out, int out_size, void* d_ws, size_t ws_size,
                              hipStream_t stream) {
  const float* xs  = (const float*)d_in[0];
  const float* xv  = (const float*)d_in[1];
  const int*   ei  = (const int*)  d_in[2];
  const float* eas = (const float*)d_in[3];
  const float* eav = (const float*)d_in[4];
  const int*   sp  = (const int*)  d_in[5];
  const float *qp[6], *kp[6], *vp[6], *op[6];
  for (int i = 0; i < 6; ++i) {
    qp[i] = (const float*)d_in[6 + i];
    kp[i] = (const float*)d_in[12 + i];
    vp[i] = (const float*)d_in[18 + i];
    op[i] = (const float*)d_in[24 + i];
  }
  const float* ewh = (const float*)d_in[30];
  const float* esw = (const float*)d_in[31];
  const float* esb = (const float*)d_in[32];
  const float* emb = (const float*)d_in[33];

  // workspace carve-up (~38 MB total; all L2-resident on MI455X)
  size_t NN = (size_t)N_NODES * N_NODES;
  char* w = (char*)d_ws;
  float*    bias  = (float*)w;    w += (size_t)NH * NN * sizeof(float);          // 16.8 MB
  _Float16* Qp    = (_Float16*)w; w += (size_t)NPLANE * N_NODES * KP * 2;        // 4 MB
  _Float16* Kp    = (_Float16*)w; w += (size_t)NPLANE * N_NODES * KP * 2;        // 4 MB
  _Float16* Vt    = (_Float16*)w; w += (size_t)NPLANE * KP * N_NODES * 2;        // 4 MB
  float*    Of    = (float*)w;    w += (size_t)NPLANE * N_NODES * KP * 4;        // 8 MB
  float*    partM = (float*)w;    w += (size_t)NPLANE * N_NODES * 4;
  float*    partS = (float*)w;    w += (size_t)NPLANE * N_NODES * 4;
  float*    rowM  = (float*)w;    w += (size_t)N_NODES * 4;
  float*    rowI  = (float*)w;    w += (size_t)N_NODES * 4;

  bias_build_kernel<<<NN / 256, 256, 0, stream>>>(sp, emb, bias);
  edge_bias_kernel<<<E_EDGES / 256, 256, 0, stream>>>(ei, eas, eav, ewh, esw, esb,
                                                      sp, emb, bias);
  gvp_pack_kernel<<<N_NODES * C_CONF, 128, 0, stream>>>(xs, xv,
      qp[0], qp[1], qp[2], qp[3], qp[4], qp[5], Qp, 0);
  gvp_pack_kernel<<<N_NODES * C_CONF, 128, 0, stream>>>(xs, xv,
      kp[0], kp[1], kp[2], kp[3], kp[4], kp[5], Kp, 1);
  gvp_pack_kernel<<<N_NODES * C_CONF, 128, 0, stream>>>(xs, xv,
      vp[0], vp[1], vp[2], vp[3], vp[4], vp[5], Vt, 2);
  attn_stats_kernel<<<dim3(N_NODES / 16, NPLANE), 32, 0, stream>>>(Qp, Kp, bias,
                                                                   partM, partS);
  softmax_combine_kernel<<<N_NODES / 256, 256, 0, stream>>>(partM, partS, rowM, rowI);
  attn_av_kernel<<<dim3(N_NODES / 16, NPLANE), 32, 0, stream>>>(Qp, Kp, Vt, bias,
                                                                rowM, rowI, Of);
  gvp_out_kernel<<<N_NODES * C_CONF, 128, 0, stream>>>(Of,
      op[0], op[1], op[2], op[3], op[4], op[5], (float*)d_out);
}
